// CausalAttention_19413252178219
// MI455X (gfx1250) — matrix operand order repaired and checked by reading, not compiled, not run on hardware
//
#include <hip/hip_runtime.h>
#include <hip/hip_bf16.h>

// ---------------------------------------------------------------------------
// CDNA5 (gfx1250) causal attention with ALiBi, bf16 WMMA pipeline.
// B=1, T=4096, C=1024, H=16, HD=64.
// ---------------------------------------------------------------------------

typedef __bf16 bf16_t;
typedef __attribute__((ext_vector_type(16))) __bf16 v16bf;
typedef __attribute__((ext_vector_type(8)))  __bf16 v8bf;
typedef __attribute__((ext_vector_type(4)))  __bf16 v4bf;
typedef __attribute__((ext_vector_type(8)))  float  v8f;

#define T_DIM 4096
#define C_DIM 1024
#define H_DIM 16
#define HD_DIM 64

__device__ __forceinline__ bf16_t f2bf(float f) {
    unsigned int u = __builtin_bit_cast(unsigned int, f);
    unsigned int r = (u + 0x7FFFu + ((u >> 16) & 1u)) >> 16;
    unsigned short s = (unsigned short)r;
    return __builtin_bit_cast(bf16_t, s);
}

__device__ __forceinline__ v8f wmma_bf16(v16bf a, v16bf b, v8f c) {
    // v_wmma_f32_16x16x32_bf16: D = A(16x32) * B(32x16) + C(16x16 f32)
    return __builtin_amdgcn_wmma_f32_16x16x32_bf16(
        /*neg_a=*/false, a, /*neg_b=*/false, b,
        /*c_mod=*/(short)0, c, /*reuse_a=*/false, /*reuse_b=*/false);
}

// A-fragment (16x32, M x K) from row-major [ld] bf16 matrix.
// ISA layout: lanes 0-15 hold M=lane; elems 0..7 -> K 0..7, elems 8..15 -> K 16..23.
// lanes 16-31 hold M=lane-16; elems 0..7 -> K 8..15, elems 8..15 -> K 24..31.
__device__ __forceinline__ v16bf load_a_frag(const bf16_t* base, int ld,
                                             int row, int col, int lane) {
    int half = lane >> 4;
    int m    = lane & 15;
    const bf16_t* p = base + (row + m) * ld + col + half * 8;
    union { v8bf h[2]; v16bf v; } u;
    u.h[0] = *(const v8bf*)(p);
    u.h[1] = *(const v8bf*)(p + 16);
    return u.v;
}

// B-fragment (32x16, K x N) where logical B(k,n) = Mrow[nbase+n][kbase+k]
// (i.e. Mrow holds B transposed, row-major). lane&15 = n; elems 0..7 -> K 8*half.., elems 8..15 -> K 16+8*half...
__device__ __forceinline__ v16bf load_b_frag(const bf16_t* base, int ld,
                                             int nbase, int kbase, int lane) {
    int half = lane >> 4;
    int n    = lane & 15;
    const bf16_t* p = base + (nbase + n) * ld + kbase + half * 8;
    union { v8bf h[2]; v16bf v; } u;
    u.h[0] = *(const v8bf*)(p);
    u.h[1] = *(const v8bf*)(p + 16);
    return u.v;
}

// ---------------------------------------------------------------------------
// Stage 0: fp32 -> bf16 elementwise convert (vectorized x4)
// ---------------------------------------------------------------------------
__global__ __launch_bounds__(256) void cvt_f32_bf16_kernel(const float4* __restrict__ in,
                                                           v4bf* __restrict__ out, int n4) {
    int i = blockIdx.x * 256 + threadIdx.x;
    if (i < n4) {
        float4 f = in[i];
        v4bf o;
        o[0] = f2bf(f.x); o[1] = f2bf(f.y); o[2] = f2bf(f.z); o[3] = f2bf(f.w);
        out[i] = o;
    }
}

// ---------------------------------------------------------------------------
// Stage 0b: W[k][n] fp32 -> Wt[n][k] bf16 (LDS tiled transpose)
// ---------------------------------------------------------------------------
__global__ __launch_bounds__(256) void transpose_cvt_kernel(const float* __restrict__ W,
                                                            bf16_t* __restrict__ Wt) {
    __shared__ float tile[32][33];
    int bx = blockIdx.x * 32;  // n base
    int by = blockIdx.y * 32;  // k base
    int tx = threadIdx.x & 31;
    int ty = threadIdx.x >> 5;
#pragma unroll
    for (int i = 0; i < 32; i += 8)
        tile[ty + i][tx] = W[(by + ty + i) * C_DIM + bx + tx];
    __syncthreads();
#pragma unroll
    for (int i = 0; i < 32; i += 8)
        Wt[(bx + ty + i) * C_DIM + by + tx] = f2bf(tile[tx][ty + i]);
}

// ---------------------------------------------------------------------------
// Stage 1: projection GEMM  Y = X(bf16) @ Wt^T + bias
//   8 waves/block (2m x 4n), 64x32 tile per wave -> 128x128 block tile.
//   transposeOut==0 : out[h][t][d]      (Q, K layout)
//   transposeOut==1 : out[h*64+d][t]    (V transposed for PV B-fragments)
// ---------------------------------------------------------------------------
__global__ __launch_bounds__(256) void qkv_gemm_kernel(const bf16_t* __restrict__ X,
                                                       const bf16_t* __restrict__ Wt,
                                                       const float* __restrict__ bias,
                                                       bf16_t* __restrict__ out,
                                                       int transposeOut) {
    int wave = threadIdx.x >> 5;
    int lane = threadIdx.x & 31;
    int wm = wave & 1, wn = wave >> 1;
    int m0 = blockIdx.x * 128 + wm * 64;
    int n0 = blockIdx.y * 128 + wn * 32;

    v8f acc[4][2] = {};
    for (int k = 0; k < C_DIM; k += 32) {
        v16bf a[4], b[2];
#pragma unroll
        for (int mi = 0; mi < 4; ++mi) a[mi] = load_a_frag(X, C_DIM, m0 + mi * 16, k, lane);
#pragma unroll
        for (int nj = 0; nj < 2; ++nj) b[nj] = load_b_frag(Wt, C_DIM, n0 + nj * 16, k, lane);
#pragma unroll
        for (int mi = 0; mi < 4; ++mi)
#pragma unroll
            for (int nj = 0; nj < 2; ++nj)
                acc[mi][nj] = wmma_bf16(a[mi], b[nj], acc[mi][nj]);
    }

    int half = lane >> 4, nl = lane & 15;
#pragma unroll
    for (int mi = 0; mi < 4; ++mi) {
#pragma unroll
        for (int nj = 0; nj < 2; ++nj) {
            int n = n0 + nj * 16 + nl;
            float bv = bias[n];
#pragma unroll
            for (int v = 0; v < 8; ++v) {
                int t = m0 + mi * 16 + v + 8 * half;
                float val = acc[mi][nj][v] + bv;
                int idx;
                if (transposeOut) {
                    idx = n * T_DIM + t;                       // Vt[c][t]
                } else {
                    int h = n >> 6, d = n & 63;
                    idx = (h * T_DIM + t) * HD_DIM + d;        // Q/K [h][t][d]
                }
                out[idx] = f2bf(val);
            }
        }
    }
}

// ---------------------------------------------------------------------------
// Stage 2: flash attention. 1 wave per (32-query tile, head).
//   S = Q K^T * 0.125 + alibi, causal mask, online softmax, O += P V.
//   K/V fragments amortized over 2 query M-tiles; P through LDS.
// ---------------------------------------------------------------------------
__global__ __launch_bounds__(32) void attn_kernel(const bf16_t* __restrict__ Q,
                                                  const bf16_t* __restrict__ K,
                                                  const bf16_t* __restrict__ Vt,
                                                  bf16_t* __restrict__ O) {
    __shared__ bf16_t pbuf[2][16 * 32];

    int lane = threadIdx.x;
    int q0 = blockIdx.x * 32;
    int h  = blockIdx.y;
    int half = lane >> 4, nl = lane & 15;

    const bf16_t* Qh = Q  + (size_t)h * T_DIM * HD_DIM;
    const bf16_t* Kh = K  + (size_t)h * T_DIM * HD_DIM;
    const bf16_t* Vh = Vt + (size_t)h * HD_DIM * T_DIM;

    v16bf qa[2][2];
#pragma unroll
    for (int mi = 0; mi < 2; ++mi) {
        qa[mi][0] = load_a_frag(Qh, HD_DIM, q0 + mi * 16, 0,  lane);
        qa[mi][1] = load_a_frag(Qh, HD_DIM, q0 + mi * 16, 32, lane);
    }

    v8f o[2][4] = {};
    float mrow[2][8], lrow[2][8];
#pragma unroll
    for (int mi = 0; mi < 2; ++mi)
#pragma unroll
        for (int v = 0; v < 8; ++v) { mrow[mi][v] = -1e30f; lrow[mi][v] = 0.0f; }

    float slope = exp2f(-0.5f * (float)(h + 1));

    int nkb = ((q0 + 31) >> 5) + 1;  // key blocks of 32 covering keys [0, q0+31]
    for (int kb = 0; kb < nkb; ++kb) {
        int key0 = kb * 32;

        // Prefetch next key block's K rows (4KB) and V columns while we compute.
        {
            const char* nk = (const char*)(Kh + (size_t)(key0 + 32) * HD_DIM);
            __builtin_prefetch(nk + lane * 128, 0, 0);
            __builtin_prefetch((const char*)(Vh + (size_t)(lane * 2 + 0) * T_DIM + key0 + 32), 0, 0);
            __builtin_prefetch((const char*)(Vh + (size_t)(lane * 2 + 1) * T_DIM + key0 + 32), 0, 0);
        }

        // ---- K fragments (shared across both query M-tiles) ----
        v16bf kb0l = load_b_frag(Kh, HD_DIM, key0,      0,  lane);
        v16bf kb0h = load_b_frag(Kh, HD_DIM, key0,      32, lane);
        v16bf kb1l = load_b_frag(Kh, HD_DIM, key0 + 16, 0,  lane);
        v16bf kb1h = load_b_frag(Kh, HD_DIM, key0 + 16, 32, lane);

        // ---- S = Q K^T : per M-tile a 16x32 tile (two 16x16 accumulators) ----
        v8f s[2][2] = {};
#pragma unroll
        for (int mi = 0; mi < 2; ++mi) {
            s[mi][0] = wmma_bf16(qa[mi][0], kb0l, s[mi][0]);
            s[mi][0] = wmma_bf16(qa[mi][1], kb0h, s[mi][0]);
            s[mi][1] = wmma_bf16(qa[mi][0], kb1l, s[mi][1]);
            s[mi][1] = wmma_bf16(qa[mi][1], kb1h, s[mi][1]);
        }

        __syncthreads();  // previous P reads complete before overwrite

        // ---- scale + alibi + causal mask + online softmax ----
        int c0 = key0 + nl;
        int c1 = key0 + 16 + nl;
        float al0 = -slope * (float)(T_DIM - 1 - c0);
        float al1 = -slope * (float)(T_DIM - 1 - c1);
#pragma unroll
        for (int mi = 0; mi < 2; ++mi) {
#pragma unroll
            for (int v = 0; v < 8; ++v) {
                int row = q0 + mi * 16 + v + 8 * half;
                float e0 = s[mi][0][v] * 0.125f + al0;
                float e1 = s[mi][1][v] * 0.125f + al1;
                if (c0 > row) e0 = -1e30f;
                if (c1 > row) e1 = -1e30f;

                float mx = fmaxf(e0, e1);
#pragma unroll
                for (int d = 1; d < 16; d <<= 1) mx = fmaxf(mx, __shfl_xor(mx, d, 32));
                float mnew  = fmaxf(mrow[mi][v], mx);
                float scale = __expf(mrow[mi][v] - mnew);
                float p0 = __expf(e0 - mnew);
                float p1 = __expf(e1 - mnew);
                float rs = p0 + p1;
#pragma unroll
                for (int d = 1; d < 16; d <<= 1) rs += __shfl_xor(rs, d, 32);
                lrow[mi][v] = lrow[mi][v] * scale + rs;
                mrow[mi][v] = mnew;
#pragma unroll
                for (int t = 0; t < 4; ++t) o[mi][t][v] *= scale;

                int r = v + 8 * half;
                pbuf[mi][r * 32 + nl]      = f2bf(p0);
                pbuf[mi][r * 32 + 16 + nl] = f2bf(p1);
            }
        }
        __syncthreads();

        // ---- P: LDS -> A-layout ----
        v16bf pa[2];
#pragma unroll
        for (int mi = 0; mi < 2; ++mi) {
            const bf16_t* p = &pbuf[mi][nl * 32 + half * 8];
            union { v8bf hh[2]; v16bf v; } u;
            u.hh[0] = *(const v8bf*)(p);
            u.hh[1] = *(const v8bf*)(p + 16);
            pa[mi] = u.v;
        }

        // ---- O += P V  (V pre-transposed: Vt[d][t]; fragments shared by M-tiles) ----
#pragma unroll
        for (int t = 0; t < 4; ++t) {
            v16bf vb = load_b_frag(Vh, T_DIM, t * 16, key0, lane);
            o[0][t] = wmma_bf16(pa[0], vb, o[0][t]);
            o[1][t] = wmma_bf16(pa[1], vb, o[1][t]);
        }
    }

    // ---- epilogue: O /= l, store bf16 [t][h*64+d] for output projection ----
#pragma unroll
    for (int mi = 0; mi < 2; ++mi) {
#pragma unroll
        for (int t = 0; t < 4; ++t) {
#pragma unroll
            for (int v = 0; v < 8; ++v) {
                int row = q0 + mi * 16 + v + 8 * half;
                int c   = h * HD_DIM + t * 16 + nl;
                float val = o[mi][t][v] / lrow[mi][v];
                O[(size_t)row * C_DIM + c] = f2bf(val);
            }
        }
    }
}

// ---------------------------------------------------------------------------
// Stage 3: output projection  out = A(bf16) @ Wot^T + bo   (f32 result)
//   Same 64x32-per-wave tiling as stage 1.
// ---------------------------------------------------------------------------
__global__ __launch_bounds__(256) void out_gemm_kernel(const bf16_t* __restrict__ A,
                                                       const bf16_t* __restrict__ Wt,
                                                       const float* __restrict__ bias,
                                                       float* __restrict__ out) {
    int wave = threadIdx.x >> 5;
    int lane = threadIdx.x & 31;
    int wm = wave & 1, wn = wave >> 1;
    int m0 = blockIdx.x * 128 + wm * 64;
    int n0 = blockIdx.y * 128 + wn * 32;

    v8f acc[4][2] = {};
    for (int k = 0; k < C_DIM; k += 32) {
        v16bf a[4], b[2];
#pragma unroll
        for (int mi = 0; mi < 4; ++mi) a[mi] = load_a_frag(A, C_DIM, m0 + mi * 16, k, lane);
#pragma unroll
        for (int nj = 0; nj < 2; ++nj) b[nj] = load_b_frag(Wt, C_DIM, n0 + nj * 16, k, lane);
#pragma unroll
        for (int mi = 0; mi < 4; ++mi)
#pragma unroll
            for (int nj = 0; nj < 2; ++nj)
                acc[mi][nj] = wmma_bf16(a[mi], b[nj], acc[mi][nj]);
    }

    int half = lane >> 4, nl = lane & 15;
#pragma unroll
    for (int mi = 0; mi < 4; ++mi) {
#pragma unroll
        for (int nj = 0; nj < 2; ++nj) {
            int n = n0 + nj * 16 + nl;
            float bv = bias[n];
#pragma unroll
            for (int v = 0; v < 8; ++v) {
                int t = m0 + mi * 16 + v + 8 * half;
                out[(size_t)t * C_DIM + n] = acc[mi][nj][v] + bv;
            }
        }
    }
}

// ---------------------------------------------------------------------------
extern "C" void kernel_launch(void* const* d_in, const int* in_sizes, int n_in,
                              void* d_out, int out_size, void* d_ws, size_t ws_size,
                              hipStream_t stream) {
    const float* x  = (const float*)d_in[0];
    const float* Wq = (const float*)d_in[1];
    const float* bq = (const float*)d_in[2];
    const float* Wk = (const float*)d_in[3];
    const float* bk = (const float*)d_in[4];
    const float* Wv = (const float*)d_in[5];
    const float* bv = (const float*)d_in[6];
    const float* Wo = (const float*)d_in[7];
    const float* bo = (const float*)d_in[8];
    float* out = (float*)d_out;

    char* ws = (char*)d_ws;
    const size_t SZ_XB = (size_t)T_DIM * C_DIM * sizeof(bf16_t);  // 8 MiB
    const size_t SZ_W  = (size_t)C_DIM * C_DIM * sizeof(bf16_t);  // 2 MiB

    bf16_t* xb  = (bf16_t*)(ws);
    bf16_t* wqt = (bf16_t*)(ws + SZ_XB);
    bf16_t* wkt = (bf16_t*)(ws + SZ_XB + 1 * SZ_W);
    bf16_t* wvt = (bf16_t*)(ws + SZ_XB + 2 * SZ_W);
    bf16_t* wot = (bf16_t*)(ws + SZ_XB + 3 * SZ_W);
    bf16_t* qb  = (bf16_t*)(ws + SZ_XB + 4 * SZ_W);
    bf16_t* kb  = (bf16_t*)(ws + 2 * SZ_XB + 4 * SZ_W);
    bf16_t* vtb = (bf16_t*)(ws + 3 * SZ_XB + 4 * SZ_W);
    bf16_t* ab  = (bf16_t*)(ws + 4 * SZ_XB + 4 * SZ_W);
    // total: 5*8MiB + 4*2MiB = 48 MiB

    // Stage 0: conversions
    int n4 = (T_DIM * C_DIM) / 4;
    cvt_f32_bf16_kernel<<<(n4 + 255) / 256, 256, 0, stream>>>((const float4*)x, (v4bf*)xb, n4);
    dim3 tg(C_DIM / 32, C_DIM / 32);
    transpose_cvt_kernel<<<tg, 256, 0, stream>>>(Wq, wqt);
    transpose_cvt_kernel<<<tg, 256, 0, stream>>>(Wk, wkt);
    transpose_cvt_kernel<<<tg, 256, 0, stream>>>(Wv, wvt);
    transpose_cvt_kernel<<<tg, 256, 0, stream>>>(Wo, wot);

    // Stage 1: Q/K/V projections (grid: 128x128 block tiles)
    dim3 gg(T_DIM / 128, C_DIM / 128);
    qkv_gemm_kernel<<<gg, 256, 0, stream>>>(xb, wqt, bq, qb, 0);
    qkv_gemm_kernel<<<gg, 256, 0, stream>>>(xb, wkt, bk, kb, 0);
    qkv_gemm_kernel<<<gg, 256, 0, stream>>>(xb, wvt, bv, vtb, 1);

    // Stage 2: attention (32 query rows per wave)
    attn_kernel<<<dim3(T_DIM / 32, H_DIM), 32, 0, stream>>>(qb, kb, vtb, ab);

    // Stage 3: output projection
    out_gemm_kernel<<<gg, 256, 0, stream>>>(ab, wot, bo, out);
}
